// GDAU_14508399526305
// MI455X (gfx1250) — matrix operand order repaired and checked by reading, not compiled, not run on hardware
//
#include <hip/hip_runtime.h>
#include <math.h>

// ---------------------------------------------------------------------------
// Shapes
// ---------------------------------------------------------------------------
constexpr int DIMC   = 192;     // channels
constexpr int NHEADS = 6;
constexpr int CHD    = 32;      // channels per head
constexpr int IMGW   = 256;
constexpr int HWN    = 65536;   // 256*256
constexpr int NBATCH = 2;

typedef __attribute__((ext_vector_type(16))) _Float16 v16h;
typedef __attribute__((ext_vector_type(8)))  float    v8f;

// ---------------------------------------------------------------------------
// LayerNorm (channels_first, biased variance, eps=1e-6). One thread = 1 pixel.
// x is ~100MB -> fully L2-resident (192MB L2), so the second pass re-read is
// cheap; avoids holding 192 values in registers.
// ---------------------------------------------------------------------------
__global__ void ln_kernel(const float* __restrict__ x, const float* __restrict__ w,
                          const float* __restrict__ b, float* __restrict__ y) {
  int s  = blockIdx.x * blockDim.x + threadIdx.x;
  int bi = blockIdx.y;
  const float* xb = x + (size_t)bi * DIMC * HWN + s;
  float*       yb = y + (size_t)bi * DIMC * HWN + s;
  float sum = 0.f, ss = 0.f;
  #pragma unroll 4
  for (int c = 0; c < DIMC; ++c) { float v = xb[(size_t)c * HWN]; sum += v; ss += v * v; }
  float mean = sum * (1.0f / DIMC);
  float var  = ss * (1.0f / DIMC) - mean * mean;
  float inv  = rsqrtf(var + 1e-6f);
  #pragma unroll 4
  for (int c = 0; c < DIMC; ++c) {
    float v = xb[(size_t)c * HWN];
    yb[(size_t)c * HWN] = w[c] * ((v - mean) * inv) + b[c];
  }
}

// ---------------------------------------------------------------------------
// conv1x1 as WMMA GEMM:  Y[o, p] = sum_c W[o,c] * X[c, p] (+ bias, + residual)
// Block: 256 threads (8 waves), tile = all 192 out-channels x 128 pixels.
// X tile staged to LDS transposed as f16; W streamed in K-slabs of 32.
// Each wave owns one 16-pixel N tile and all 12 M tiles (12 x v8f accums),
// issuing v_wmma_f32_16x16x32_f16 per (m-tile, k-step).
// ---------------------------------------------------------------------------
constexpr int NTILE = 128;
constexpr int XLS   = 196;   // halves stride for x tile (pad vs 192)
constexpr int WLS   = 36;    // halves stride for w slab (pad vs 32)

__global__ void __launch_bounds__(256) gemm192_wmma_kernel(
    const float* __restrict__ X, const float* __restrict__ Wm,
    const float* __restrict__ bias, const float* __restrict__ resid,
    float* __restrict__ Y) {
  __shared__ _Float16 xlds[NTILE * XLS];   // 50176 B
  __shared__ _Float16 wslab[DIMC * WLS];   // 13824 B  (total 64000 B)
  int tid = threadIdx.x;
  int bi  = blockIdx.y;
  int s0  = blockIdx.x * NTILE;
  const float* Xb = X + (size_t)bi * DIMC * HWN;

  // Stage X tile transposed: xlds[n][c] = X[c][s0+n]  (coalesced global reads)
  for (int i = tid; i < DIMC * NTILE; i += 256) {
    int c = i / NTILE, n = i % NTILE;
    xlds[n * XLS + c] = (_Float16)Xb[(size_t)c * HWN + s0 + n];
  }

  int wave = tid >> 5, lane = tid & 31;
  int hh = lane >> 4, l16 = lane & 15;   // half-wave select, lane-in-16
  int nb = wave * 16;                    // this wave's pixel sub-tile
  v8f acc[12] = {};
  const _Float16* xrow = &xlds[(nb + l16) * XLS];

  for (int ks = 0; ks < 6; ++ks) {       // K = 192 in steps of 32
    __syncthreads();                     // (first iter: covers X staging)
    for (int i = tid; i < DIMC * 32; i += 256) {
      int m = i >> 5, kk = i & 31;
      wslab[m * WLS + kk] = (_Float16)Wm[m * DIMC + ks * 32 + kk];
    }
    __syncthreads();

    // B fragment (32x16, K x N): half j of lane = B[K = 16*hh + j][N = l16]
    v16h bf;
    #pragma unroll
    for (int j = 0; j < 16; ++j) bf[j] = xrow[ks * 32 + hh * 16 + j];

    #pragma unroll
    for (int mt = 0; mt < 12; ++mt) {
      // A fragment (16x32, M x K): half j of lane = A[M = l16][K = 16*hh + j]
      const _Float16* wr = &wslab[(mt * 16 + l16) * WLS];
      v16h a;
      #pragma unroll
      for (int i = 0; i < 8; ++i) {
        int kb = 2 * i + 16 * hh;
        a[2 * i]     = wr[kb];
        a[2 * i + 1] = wr[kb + 1];
      }
      acc[mt] = __builtin_amdgcn_wmma_f32_16x16x32_f16(
          false, a, false, bf, (short)0, acc[mt], false, false);
    }
  }

  // C/D layout: VGPR r of lane -> M = r + 8*hh, N = l16
  int ncol = s0 + nb + l16;
  #pragma unroll
  for (int mt = 0; mt < 12; ++mt) {
    #pragma unroll
    for (int r = 0; r < 8; ++r) {
      int mr = mt * 16 + r + 8 * hh;
      size_t off = ((size_t)bi * DIMC + mr) * HWN + ncol;
      float val = acc[mt][r] + bias[mr];
      if (resid) val += resid[off];
      Y[off] = val;
    }
  }
}

// ---------------------------------------------------------------------------
// Depthwise 3x3, padding SAME (cross-correlation, matches lax/torch).
// ---------------------------------------------------------------------------
__global__ void dw_kernel(const float* __restrict__ in, const float* __restrict__ kern,
                          const float* __restrict__ bias, float* __restrict__ out) {
  int s  = blockIdx.x * blockDim.x + threadIdx.x;
  int c  = blockIdx.y;
  int bi = blockIdx.z;
  int y = s >> 8, x = s & 255;
  const float* ib = in + ((size_t)bi * DIMC + c) * HWN;
  const float* kc = kern + c * 9;
  float acc = bias[c];
  #pragma unroll
  for (int dy = -1; dy <= 1; ++dy) {
    int yy = y + dy;
    if (yy < 0 || yy >= IMGW) continue;
    #pragma unroll
    for (int dx = -1; dx <= 1; ++dx) {
      int xx = x + dx;
      if (xx < 0 || xx >= IMGW) continue;
      acc += kc[(dy + 1) * 3 + (dx + 1)] * ib[yy * IMGW + xx];
    }
  }
  out[((size_t)bi * DIMC + c) * HWN + s] = acc;
}

// ---------------------------------------------------------------------------
// L2 norms over the spatial axis: one workgroup per (b, channel) row.
// blockIdx.y: 0 -> q/nq, 1 -> k/nk.
// ---------------------------------------------------------------------------
__global__ void rownorm_kernel(const float* __restrict__ q, const float* __restrict__ k,
                               float* __restrict__ nq, float* __restrict__ nk) {
  __shared__ float red[256];
  int row = blockIdx.x;  // b*192 + channel
  const float* src = (blockIdx.y == 0 ? q : k) + (size_t)row * HWN;
  float ss = 0.f;
  for (int i = threadIdx.x; i < HWN; i += 256) { float v = src[i]; ss += v * v; }
  red[threadIdx.x] = ss;
  __syncthreads();
  for (int o = 128; o > 0; o >>= 1) {
    if (threadIdx.x < o) red[threadIdx.x] += red[threadIdx.x + o];
    __syncthreads();
  }
  if (threadIdx.x == 0) (blockIdx.y == 0 ? nq : nk)[row] = sqrtf(red[0]);
}

// ---------------------------------------------------------------------------
// Channel-attention logits: raw[bh][cq][ck] = sum_s q[cq,s]*k[ck,s]
// Partial sums over 2048-pixel chunks, f32 atomicAdd into zeroed buffer.
// ---------------------------------------------------------------------------
constexpr int ACHUNK = 2048;
constexpr int ATILE  = 64;

__global__ void attn_zero_kernel(float* __restrict__ a) {
  int i = blockIdx.x * 256 + threadIdx.x;
  if (i < NBATCH * NHEADS * CHD * CHD) a[i] = 0.f;
}

__global__ void attn_partial_kernel(const float* __restrict__ q, const float* __restrict__ k,
                                    float* __restrict__ attn) {
  __shared__ float qs[CHD][ATILE + 1];
  __shared__ float ks[CHD][ATILE + 1];
  int bh = blockIdx.y;                                  // b*6 + h
  const float* qb = q + (size_t)bh * CHD * HWN;         // row (b*192+h*32+cq)
  const float* kb = k + (size_t)bh * CHD * HWN;
  int s0  = blockIdx.x * ACHUNK;
  int ck  = threadIdx.x & 31;
  int cq0 = threadIdx.x >> 5;                           // 0..7
  float acc[4] = {0.f, 0.f, 0.f, 0.f};
  for (int t = 0; t < ACHUNK; t += ATILE) {
    for (int i = threadIdx.x; i < CHD * ATILE; i += 256) {
      int r = i / ATILE, col = i % ATILE;
      qs[r][col] = qb[(size_t)r * HWN + s0 + t + col];
      ks[r][col] = kb[(size_t)r * HWN + s0 + t + col];
    }
    __syncthreads();
    #pragma unroll
    for (int p = 0; p < 4; ++p) {
      int cq = cq0 + 8 * p;
      float a = 0.f;
      for (int s = 0; s < ATILE; ++s) a += qs[cq][s] * ks[ck][s];
      acc[p] += a;
    }
    __syncthreads();
  }
  #pragma unroll
  for (int p = 0; p < 4; ++p)
    atomicAdd(&attn[((size_t)bh * CHD + cq0 + 8 * p) * CHD + ck], acc[p]);
}

// ---------------------------------------------------------------------------
// Softmax over ck, folding in temperature and clamped q/k norms:
//   logit = raw * temp[h] / (max(|q|,1e-12) * max(|k|,1e-12))
// One thread per row (384 rows total).
// ---------------------------------------------------------------------------
__global__ void softmax_kernel(float* __restrict__ attn, const float* __restrict__ nq,
                               const float* __restrict__ nk, const float* __restrict__ temp) {
  int row = blockIdx.x * blockDim.x + threadIdx.x;      // b*192 + h*32 + cq
  if (row >= NBATCH * DIMC) return;
  int h = (row / CHD) % NHEADS;
  float* ar = attn + (size_t)row * CHD;
  const float* nkr = nk + (row / CHD) * CHD;            // base b*192 + h*32
  float qn = fmaxf(nq[row], 1e-12f);
  float t  = temp[h];
  float vals[CHD];
  float mx = -3.4e38f;
  #pragma unroll
  for (int j = 0; j < CHD; ++j) {
    float kn = fmaxf(nkr[j], 1e-12f);
    float v = ar[j] * t / (qn * kn);
    vals[j] = v;
    mx = fmaxf(mx, v);
  }
  float sum = 0.f;
  #pragma unroll
  for (int j = 0; j < CHD; ++j) { float e = expf(vals[j] - mx); vals[j] = e; sum += e; }
  float inv = 1.0f / sum;
  #pragma unroll
  for (int j = 0; j < CHD; ++j) ar[j] = vals[j] * inv;
}

// ---------------------------------------------------------------------------
// x2 = (attn @ v) * gelu_exact(x_ffn). One thread = 1 pixel; attn in LDS.
// ---------------------------------------------------------------------------
__global__ void gated_kernel(const float* __restrict__ v, const float* __restrict__ xffn,
                             const float* __restrict__ attn, float* __restrict__ x2) {
  __shared__ float at[NHEADS * CHD * CHD];              // 24 KB
  int bi = blockIdx.y;
  for (int i = threadIdx.x; i < NHEADS * CHD * CHD; i += 256)
    at[i] = attn[(size_t)bi * NHEADS * CHD * CHD + i];
  __syncthreads();
  int s = blockIdx.x * 256 + threadIdx.x;
  const float* vb = v    + (size_t)bi * DIMC * HWN + s;
  const float* fb = xffn + (size_t)bi * DIMC * HWN + s;
  float*       ob = x2   + (size_t)bi * DIMC * HWN + s;
  for (int h = 0; h < NHEADS; ++h) {
    float vr[CHD];
    #pragma unroll
    for (int d = 0; d < CHD; ++d) vr[d] = vb[(size_t)(h * CHD + d) * HWN];
    const float* ah = at + h * CHD * CHD;
    #pragma unroll 4
    for (int c = 0; c < CHD; ++c) {
      float o = 0.f;
      #pragma unroll
      for (int d = 0; d < CHD; ++d) o += ah[c * CHD + d] * vr[d];
      float g  = fb[(size_t)(h * CHD + c) * HWN];
      float ge = 0.5f * g * (1.0f + erff(g * 0.70710678118654752f));   // exact gelu
      ob[(size_t)(h * CHD + c) * HWN] = o * ge;
    }
  }
}

// ---------------------------------------------------------------------------
// Launcher
// ---------------------------------------------------------------------------
extern "C" void kernel_launch(void* const* d_in, const int* in_sizes, int n_in,
                              void* d_out, int out_size, void* d_ws, size_t ws_size,
                              hipStream_t stream) {
  const float* x     = (const float*)d_in[0];
  const float* ln_w  = (const float*)d_in[1];
  const float* ln_b  = (const float*)d_in[2];
  const float* w_qkv = (const float*)d_in[3];
  const float* b_qkv = (const float*)d_in[4];
  const float* k_qdw = (const float*)d_in[5];
  const float* b_qdw = (const float*)d_in[6];
  const float* k_kdw = (const float*)d_in[7];
  const float* b_kdw = (const float*)d_in[8];
  const float* k_vdw = (const float*)d_in[9];
  const float* b_vdw = (const float*)d_in[10];
  const float* temperature = (const float*)d_in[11];
  const float* w_in  = (const float*)d_in[12];
  const float* b_in  = (const float*)d_in[13];
  const float* k_dw  = (const float*)d_in[14];
  const float* b_dw  = (const float*)d_in[15];
  const float* w_out = (const float*)d_in[16];
  const float* b_out = (const float*)d_in[17];
  float* out = (float*)d_out;

  // Workspace carve-up (attn/norms small region, then 6 big activation bufs,
  // two of which are reused once their producer's consumers are done).
  char*  ws   = (char*)d_ws;
  float* attn = (float*)ws;                             // 12288 f32
  float* nq   = (float*)(ws + 12288 * 4);               // 384 f32
  float* nk   = (float*)(ws + 12288 * 4 + 384 * 4);     // 384 f32
  const size_t BIG = (size_t)NBATCH * DIMC * HWN * sizeof(float);
  char* big = ws + (1u << 20);
  float* xn   = (float*)(big + 0 * BIG);
  float* qkv  = (float*)(big + 1 * BIG);
  float* q    = (float*)(big + 2 * BIG);
  float* k    = (float*)(big + 3 * BIG);
  float* v    = (float*)(big + 4 * BIG);
  float* x1   = (float*)(big + 5 * BIG);
  float* xffn = xn;    // xn dead after both input GEMMs
  float* x2   = qkv;   // qkv dead after q/k/v dwconvs

  dim3 blk(256);
  ln_kernel<<<dim3(HWN / 256, NBATCH), blk, 0, stream>>>(x, ln_w, ln_b, xn);
  gemm192_wmma_kernel<<<dim3(HWN / NTILE, NBATCH), blk, 0, stream>>>(xn, w_qkv, b_qkv, nullptr, qkv);
  gemm192_wmma_kernel<<<dim3(HWN / NTILE, NBATCH), blk, 0, stream>>>(xn, w_in, b_in, nullptr, x1);
  dw_kernel<<<dim3(HWN / 256, DIMC, NBATCH), blk, 0, stream>>>(qkv, k_qdw, b_qdw, q);
  dw_kernel<<<dim3(HWN / 256, DIMC, NBATCH), blk, 0, stream>>>(qkv, k_kdw, b_kdw, k);
  dw_kernel<<<dim3(HWN / 256, DIMC, NBATCH), blk, 0, stream>>>(qkv, k_vdw, b_vdw, v);
  dw_kernel<<<dim3(HWN / 256, DIMC, NBATCH), blk, 0, stream>>>(x1, k_dw, b_dw, xffn);
  rownorm_kernel<<<dim3(NBATCH * DIMC, 2), blk, 0, stream>>>(q, k, nq, nk);
  attn_zero_kernel<<<dim3(48), blk, 0, stream>>>(attn);
  attn_partial_kernel<<<dim3(HWN / ACHUNK, NBATCH * NHEADS), blk, 0, stream>>>(q, k, attn);
  softmax_kernel<<<dim3(NBATCH), dim3(DIMC), 0, stream>>>(attn, nq, nk, temperature);
  gated_kernel<<<dim3(HWN / 256, NBATCH), blk, 0, stream>>>(v, xffn, attn, x2);
  gemm192_wmma_kernel<<<dim3(HWN / NTILE, NBATCH), blk, 0, stream>>>(x2, w_out, b_out, x, out);
}
